// MoELayer_65283502899666
// MI455X (gfx1250) — matrix operand order repaired and checked by reading, not compiled, not run on hardware
//
#include <hip/hip_runtime.h>
#include <hip/hip_bf16.h>

// ---------------- CDNA5 WMMA types ----------------
typedef __attribute__((ext_vector_type(16))) __bf16 v16bf;
typedef __attribute__((ext_vector_type(8)))  float  v8f;

union BFr { unsigned int u[8]; v16bf v; };   // 16 bf16 = 8 dwords
union Fr8 { float f[8]; v8f v; };

__device__ __forceinline__ unsigned short f2bf(float f) {
    unsigned int u = __float_as_uint(f);
    unsigned int r = u + 0x7FFFu + ((u >> 16) & 1u);   // round-to-nearest-even
    return (unsigned short)(r >> 16);
}
__device__ __forceinline__ unsigned int pack2(unsigned short lo, unsigned short hi) {
    return (unsigned int)lo | ((unsigned int)hi << 16);
}
__device__ __forceinline__ v8f wmma_bf16(v16bf a, v16bf b, v8f c) {
    return __builtin_amdgcn_wmma_f32_16x16x32_bf16(false, a, false, b, (short)0, c, false, false);
}

#define HH 768
#define II 2048
#define NE 8
#define NT 1024
#define XS 776    /* padded LDS stride for X (bf16 elems): conflict-free b128 */
#define HS 1032   /* padded LDS stride for H chunk */

// ---------------- Router: logits -> top-2 -> e_sel = max index; scatter ----------------
__global__ __launch_bounds__(256) void moe_router(const float* __restrict__ x,
                                                  const float* __restrict__ rw,
                                                  int* __restrict__ cnt,
                                                  int* __restrict__ lists) {
    const int wave = threadIdx.x >> 5;
    const int lane = threadIdx.x & 31;
    const int t = blockIdx.x * 8 + wave;
    float acc[NE];
#pragma unroll
    for (int e = 0; e < NE; ++e) acc[e] = 0.f;
    for (int h = lane; h < HH; h += 32) {
        float xv = x[t * HH + h];
#pragma unroll
        for (int e = 0; e < NE; ++e) acc[e] += xv * rw[e * HH + h];
    }
#pragma unroll
    for (int e = 0; e < NE; ++e) {
#pragma unroll
        for (int off = 16; off > 0; off >>= 1)
            acc[e] += __shfl_xor(acc[e], off, 32);
    }
    if (lane == 0) {
        int i1 = 0; float v1 = acc[0];
#pragma unroll
        for (int e = 1; e < NE; ++e) if (acc[e] > v1) { v1 = acc[e]; i1 = e; }
        int i2 = -1; float v2 = -3.4e38f;
#pragma unroll
        for (int e = 0; e < NE; ++e) if (e != i1 && acc[e] > v2) { v2 = acc[e]; i2 = e; }
        int es = (i1 > i2) ? i1 : i2;          // last-write-wins == highest index in top-2
        int pos = atomicAdd(&cnt[es], 1);
        lists[es * NT + pos] = t;
    }
}

// ---------------- Weight pre-pack: fp32 [E,K,N] -> bf16 WMMA B-fragment layout ----------
// Fragment (e, kb, nb) = 32x16 tile, stored as 256 dwords: dword d = lane*8 + j holds
// bf16 pair K = kb*32 + (lane>>4)*16 + 2j (lo), +1 (hi), column n = nb*16 + (lane&15).
// So in the GEMM, lane `l` reads its 16 bf16 as 2 contiguous uint4 at frag + l*2.
template<int K, int N>
__global__ __launch_bounds__(256) void pack_w(const float* __restrict__ w,
                                              unsigned int* __restrict__ wp) {
    const int KB = K / 32, NB = N / 16;
    size_t tid = (size_t)blockIdx.x * 256 + threadIdx.x;
    const size_t perE = (size_t)KB * NB * 256;
    int e  = (int)(tid / perE);
    size_t r = tid - (size_t)e * perE;
    int kb = (int)(r / ((size_t)NB * 256));
    size_t r2 = r - (size_t)kb * NB * 256;
    int nb = (int)(r2 >> 8);
    int d  = (int)(r2 & 255);
    int lane = d >> 3, j = d & 7;
    int k = kb * 32 + (lane >> 4) * 16 + 2 * j;
    int n = nb * 16 + (lane & 15);
    const float* src = w + ((size_t)e * K + k) * N + n;
    wp[tid] = pack2(f2bf(src[0]), f2bf(src[N]));
}

// ---------------- FFN: per (expert, 16-token tile): out = silu(X W1e) W2e ----------------
template<bool PACKED>
__global__ __launch_bounds__(256) void moe_ffn(const float* __restrict__ x,
                                               const float* __restrict__ w1,
                                               const float* __restrict__ w2,
                                               const uint4* __restrict__ w1p,
                                               const uint4* __restrict__ w2p,
                                               const int* __restrict__ cnt,
                                               const int* __restrict__ lists,
                                               float* __restrict__ out) {
    const int e    = blockIdx.y;
    const int tile = blockIdx.x;
    const int nTok = cnt[e];
    if (tile * 16 >= nTok) return;                 // uniform early exit

    __shared__ unsigned short Xbf[16 * XS];
    __shared__ unsigned short Hbf[16 * HS];
    __shared__ int toks[16];

    const int tid = threadIdx.x;
    if (tid < 16) {
        int s = tile * 16 + tid;
        toks[tid] = (s < nTok) ? lists[e * NT + s] : -1;
    }
    __syncthreads();

    for (int idx = tid; idx < 16 * HH; idx += 256) {
        int r = idx / HH, c = idx - r * HH;
        int tk = toks[r];
        float v = (tk >= 0) ? x[(size_t)tk * HH + c] : 0.f;
        Xbf[r * XS + c] = f2bf(v);
    }
    __syncthreads();

    const float* w1e = w1 + (size_t)e * HH * II;
    const float* w2e = w2 + (size_t)e * II * HH;

    const int wv    = tid >> 5;
    const int lane  = tid & 31;
    const int lrow  = lane & 15;
    const int lhalf = lane >> 4;
    const int koffA = lhalf * 16;
    const int koffB = lhalf * 16;

    Fr8 acc2[6];
#pragma unroll
    for (int i = 0; i < 6; ++i)
#pragma unroll
        for (int j = 0; j < 8; ++j) acc2[i].f[j] = 0.f;

    for (int kc = 0; kc < 2; ++kc) {               // two 1024-wide halves of I
        // ---- Phase A: Hchunk = silu(X @ W1e[:, kc*1024 +: 1024]) ----
        for (int nt = 0; nt < 8; ++nt) {
            const int n0 = (wv * 8 + nt) * 16;                 // col within chunk
            const int gnb = (kc * 1024 + n0) >> 4;             // global nb for W1p
            Fr8 acc;
#pragma unroll
            for (int j = 0; j < 8; ++j) acc.f[j] = 0.f;
            for (int k0 = 0; k0 < HH; k0 += 32) {
                BFr a;
                uint4 a0 = *(const uint4*)&Xbf[lrow * XS + k0 + koffA];
                uint4 a1 = *(const uint4*)&Xbf[lrow * XS + k0 + 8 + koffA];
                a.u[0] = a0.x; a.u[1] = a0.y; a.u[2] = a0.z; a.u[3] = a0.w;
                a.u[4] = a1.x; a.u[5] = a1.y; a.u[6] = a1.z; a.u[7] = a1.w;
                BFr b;
                if (PACKED) {
                    const uint4* bp = w1p +
                        ((((size_t)e * (HH/32) + (k0 >> 5)) * (II/16) + gnb) << 6) + lane * 2;
                    if (k0 + 32 < HH) __builtin_prefetch(bp + ((size_t)(II/16) << 6), 0, 1);
                    uint4 b0 = bp[0], b1 = bp[1];
                    b.u[0] = b0.x; b.u[1] = b0.y; b.u[2] = b0.z; b.u[3] = b0.w;
                    b.u[4] = b1.x; b.u[5] = b1.y; b.u[6] = b1.z; b.u[7] = b1.w;
                } else {
                    const float* bp = w1e + (size_t)(k0 + koffB) * II + (gnb << 4) + lrow;
                    if (k0 + 32 < HH) __builtin_prefetch(bp + (size_t)32 * II, 0, 1);
#pragma unroll
                    for (int j = 0; j < 8; ++j) {
                        float f0 = bp[0], f1 = bp[II];
                        bp += 2 * II;
                        b.u[j] = pack2(f2bf(f0), f2bf(f1));
                    }
                }
                acc.v = wmma_bf16(a.v, b.v, acc.v);
            }
#pragma unroll
            for (int r = 0; r < 8; ++r) {
                float hv = acc.f[r];
                hv = hv / (1.f + __expf(-hv));
                int M = r + 8 * lhalf;
                Hbf[M * HS + n0 + lrow] = f2bf(hv);
            }
        }
        __syncthreads();
        // ---- Phase B: acc2 += Hchunk @ W2e[kc*1024 +: 1024, :] ----
        for (int nt = 0; nt < 6; ++nt) {
            const int n0 = (wv * 6 + nt) * 16;
            for (int k0 = 0; k0 < 1024; k0 += 32) {
                BFr a;
                uint4 a0 = *(const uint4*)&Hbf[lrow * HS + k0 + koffA];
                uint4 a1 = *(const uint4*)&Hbf[lrow * HS + k0 + 8 + koffA];
                a.u[0] = a0.x; a.u[1] = a0.y; a.u[2] = a0.z; a.u[3] = a0.w;
                a.u[4] = a1.x; a.u[5] = a1.y; a.u[6] = a1.z; a.u[7] = a1.w;
                BFr b;
                if (PACKED) {
                    const uint4* bp = w2p +
                        ((((size_t)e * (II/32) + ((kc * 1024 + k0) >> 5)) * (HH/16) + (n0 >> 4)) << 6)
                        + lane * 2;
                    if (k0 + 32 < 1024) __builtin_prefetch(bp + ((size_t)(HH/16) << 6), 0, 1);
                    uint4 b0 = bp[0], b1 = bp[1];
                    b.u[0] = b0.x; b.u[1] = b0.y; b.u[2] = b0.z; b.u[3] = b0.w;
                    b.u[4] = b1.x; b.u[5] = b1.y; b.u[6] = b1.z; b.u[7] = b1.w;
                } else {
                    const float* bp = w2e + (size_t)(kc * 1024 + k0 + koffB) * HH + n0 + lrow;
                    if (k0 + 32 < 1024) __builtin_prefetch(bp + (size_t)32 * HH, 0, 1);
#pragma unroll
                    for (int j = 0; j < 8; ++j) {
                        float f0 = bp[0], f1 = bp[HH];
                        bp += 2 * HH;
                        b.u[j] = pack2(f2bf(f0), f2bf(f1));
                    }
                }
                acc2[nt].v = wmma_bf16(a.v, b.v, acc2[nt].v);
            }
        }
        __syncthreads();
    }

    for (int nt = 0; nt < 6; ++nt) {
        const int n0 = (wv * 6 + nt) * 16;
#pragma unroll
        for (int r = 0; r < 8; ++r) {
            int M = r + 8 * lhalf;
            int tk = toks[M];
            if (tk >= 0) out[(size_t)tk * HH + n0 + lrow] = acc2[nt].f[r];
        }
    }
}

extern "C" void kernel_launch(void* const* d_in, const int* in_sizes, int n_in,
                              void* d_out, int out_size, void* d_ws, size_t ws_size,
                              hipStream_t stream) {
    (void)in_sizes; (void)n_in; (void)out_size;
    const float* x  = (const float*)d_in[0];
    const float* rw = (const float*)d_in[1];
    const float* w1 = (const float*)d_in[2];
    const float* w2 = (const float*)d_in[3];
    float* out = (float*)d_out;

    int* cnt   = (int*)d_ws;                                   // 8 counters (32 B)
    int* lists = (int*)((char*)d_ws + 32);                     // 8 x 1024 ids (32 KB)
    const size_t W1P_BYTES = (size_t)NE * HH * II * 2;         // 25.2 MB
    const size_t W2P_BYTES = (size_t)NE * II * HH * 2;         // 25.2 MB
    const size_t PACK_OFF1 = 32 + (size_t)NE * NT * 4;         // 32800 (16B aligned)
    const size_t PACK_OFF2 = PACK_OFF1 + W1P_BYTES;
    const size_t NEEDED    = PACK_OFF2 + W2P_BYTES;
    unsigned int* w1p = (unsigned int*)((char*)d_ws + PACK_OFF1);
    unsigned int* w2p = (unsigned int*)((char*)d_ws + PACK_OFF2);

    hipMemsetAsync(d_ws, 0, 32, stream);
    moe_router<<<dim3(128), dim3(256), 0, stream>>>(x, rw, cnt, lists);

    if (ws_size >= NEEDED) {
        // 8*24*128*256 = 8*64*48*256 = 6,291,456 dwords each -> 24576 blocks
        pack_w<HH, II><<<dim3(24576), dim3(256), 0, stream>>>(w1, w1p);
        pack_w<II, HH><<<dim3(24576), dim3(256), 0, stream>>>(w2, w2p);
        moe_ffn<true><<<dim3(64, NE), dim3(256), 0, stream>>>(
            x, w1, w2, (const uint4*)w1p, (const uint4*)w2p, cnt, lists, out);
    } else {
        moe_ffn<false><<<dim3(64, NE), dim3(256), 0, stream>>>(
            x, w1, w2, nullptr, nullptr, cnt, lists, out);
    }
}
